// PPOSNN_29300266893850
// MI455X (gfx1250) — matrix operand order repaired and checked by reading, not compiled, not run on hardware
//
#include <hip/hip_runtime.h>
#include <cstddef>

// ---------------------------------------------------------------------------
// PPOSNN forward on MI455X (gfx1250, wave32, WMMA).
//
// B=2048, S=2048, H=256, A=16, SEQ=10.
//   1) encode_planes_kernel: deterministic 10-step spike trains stored as
//      per-timestep BITPLANES: planes[t][b][k/32] over K=4096 (pos half then
//      neg half). 10 MB, L2-resident; one b32 supplies 32 K-bits per lane.
//   2) pack_frag_kernel: f32 weights -> bf16 hi + bf16 lo residual,
//      pre-swizzled into the v_wmma 32x16 B-fragment lane layout.
//   3) snn_scan_kernel: 64 blocks x 256 threads; each block owns 32 batch
//      rows + full H=256, so the whole recurrent scan is block-local.
//      Per wave: 2 M-tiles x 2 N-tiles -> each B-fragment feeds 2 WMMAs
//      (halves L2 traffic). A-fragments expanded from bitplanes via a
//      16-entry LDS LUT (nibble -> 2 packed bf16 dwords) to keep the VALU
//      pipe shorter than the XDL pipe.
// Workspace: ~14.3 MB.
// ---------------------------------------------------------------------------

typedef __attribute__((ext_vector_type(16))) __bf16        v16bf;
typedef __attribute__((ext_vector_type(8)))  float         v8f;
typedef __attribute__((ext_vector_type(8)))  unsigned int  u32x8;
typedef __attribute__((ext_vector_type(4)))  unsigned int  u32x4;

#define BDIM   2048
#define SDIM   2048
#define HDIM   256
#define ADIM   16
#define KIN    4096   // 2*S
#define KWORDS 128    // KIN/32 bitplane words per row
#define SEQ    10
#define ROWS   32     // batch rows per block (2 M-tiles)

// ---------------- bf16 helpers (RNE split of f32 into hi+lo) ----------------
static __device__ inline unsigned short f32_to_bf16(float f) {
  unsigned u = __builtin_bit_cast(unsigned, f);
  unsigned r = 0x7FFFu + ((u >> 16) & 1u);
  return (unsigned short)((u + r) >> 16);
}
static __device__ inline float bf16_to_f32(unsigned short h) {
  return __builtin_bit_cast(float, ((unsigned)h) << 16);
}

// ---------------------- spike bitplane encoder -------------------------------
// One thread per (b, 32-element word group). planes[t][b][w]:
//   w in [0,64)  : pos half, bit j = spike of relu(+50*x[b, w*32+j]) at step t
//   w in [64,128): neg half, bit j = spike of relu(-50*x[b, (w-64)*32+j])
__global__ __launch_bounds__(256) void encode_planes_kernel(
    const float* __restrict__ x, unsigned int* __restrict__ planes, int nWords) {
  int idx = blockIdx.x * 256 + threadIdx.x;
  if (idx >= nWords) return;
  int b  = idx >> 6;
  int wg = idx & 63;
  unsigned posw[SEQ], negw[SEQ];
#pragma unroll
  for (int t = 0; t < SEQ; ++t) { posw[t] = 0u; negw[t] = 0u; }
  const float* xp = x + (size_t)b * SDIM + wg * 32;
  for (int j = 0; j < 32; ++j) {
    float xv = xp[j];
    float c1 = fmaxf(50.0f * xv, 0.0f);
    float c2 = fmaxf(-50.0f * xv, 0.0f);
    float v1 = 0.0f, v2 = 0.0f;
#pragma unroll
    for (int t = 0; t < SEQ; ++t) {
      v1 = v1 + 0.1f * ((0.0f - v1) + c1);
      v2 = v2 + 0.1f * ((0.0f - v2) + c2);
      if (v1 - 1.0f > 0.0f) { posw[t] |= 1u << j; v1 = 0.0f; }
      if (v2 - 1.0f > 0.0f) { negw[t] |= 1u << j; v2 = 0.0f; }
    }
  }
#pragma unroll
  for (int t = 0; t < SEQ; ++t) {
    size_t base = ((size_t)t * BDIM + b) * KWORDS;
    planes[base + wg]      = posw[t];
    planes[base + 64 + wg] = negw[t];
  }
}

// ------------------- weight pre-swizzle into B fragments ---------------------
// B-fragment (16-bit, 32x16, wave32): N = ntile*16 + (lane&15);
// K = kblk*32 + (lane>>4)*8 + e (e<8) / 16 + (lane>>4)*8 + (e-8) (e>=8) (dword d packs e=2d lo16, e=2d+1 hi16).
__global__ __launch_bounds__(256) void pack_frag_kernel(
    const float* __restrict__ W /* [N][K] row-major */, int K, int nTiles,
    unsigned int* __restrict__ hi, unsigned int* __restrict__ lo, int total) {
  int idx = blockIdx.x * 256 + threadIdx.x;
  if (idx >= total) return;
  int d    = idx & 7;
  int lin  = idx >> 3;
  int lane = lin & 31;
  int q    = lin >> 5;
  int nt   = q % nTiles;
  int kblk = q / nTiles;
  int n = nt * 16 + (lane & 15);
  int k = kblk * 32 + (lane >> 4) * 8 + (d >> 2) * 16 + 2 * (d & 3);
  float w0 = W[(size_t)n * K + k];
  float w1 = W[(size_t)n * K + k + 1];
  unsigned short h0 = f32_to_bf16(w0), h1 = f32_to_bf16(w1);
  unsigned short l0 = f32_to_bf16(w0 - bf16_to_f32(h0));
  unsigned short l1 = f32_to_bf16(w1 - bf16_to_f32(h1));
  hi[idx] = (unsigned)h0 | ((unsigned)h1 << 16);
  lo[idx] = (unsigned)l0 | ((unsigned)l1 << 16);
}

// ------------------------------ fragment helpers -----------------------------
static __device__ inline size_t frag_off(int kblk, int ntile, int nTiles, int lane) {
  return ((size_t)((kblk * nTiles + ntile) * 32 + lane)) * 8;
}

static __device__ inline v8f wmma_bf16(v16bf a, u32x8 b, v8f c) {
  return __builtin_amdgcn_wmma_f32_16x16x32_bf16(
      false, a, false, __builtin_bit_cast(v16bf, b), (short)0, c, false, false);
}

// A-fragment (16-bit, 16x32): lane<16: M=lane, K = e (e<8) / e+8 (e>=8);
// lane>=16: M=lane-16, K = e+8 (e<8) / e+16 (e>=8).
static __device__ inline v16bf load_z_afrag(const unsigned short* zb,
                                            int kbase, int lrow, int half) {
  int b0 = lrow * HDIM + kbase + 8 * half;
  u32x4 x0 = *(const u32x4*)(zb + b0);
  u32x4 x1 = *(const u32x4*)(zb + b0 + 16);
  u32x8 au;
  au[0] = x0[0]; au[1] = x0[1]; au[2] = x0[2]; au[3] = x0[3];
  au[4] = x1[0]; au[5] = x1[1]; au[6] = x1[2]; au[7] = x1[3];
  return __builtin_bit_cast(v16bf, au);
}

// Expand one bitplane word (32 K-bits of this lane's row) into a bf16 {0,1}
// A-fragment via 16-entry LDS LUT: nibble -> two packed-bf16 dwords.
static __device__ inline v16bf expand_afrag(const unsigned long long* lut,
                                            unsigned w, int half) {
  int s0 = 8 * half;            // run0 K-bits at s0..s0+7
  int s1 = 16 + 8 * half;       // run1 K-bits at s1..s1+7
  unsigned long long q0 = lut[(w >> s0)       & 0xFu];
  unsigned long long q1 = lut[(w >> (s0 + 4)) & 0xFu];
  unsigned long long q2 = lut[(w >> s1)       & 0xFu];
  unsigned long long q3 = lut[(w >> (s1 + 4)) & 0xFu];
  u32x8 au;
  au[0] = (unsigned)q0; au[1] = (unsigned)(q0 >> 32);
  au[2] = (unsigned)q1; au[3] = (unsigned)(q1 >> 32);
  au[4] = (unsigned)q2; au[5] = (unsigned)(q2 >> 32);
  au[6] = (unsigned)q3; au[7] = (unsigned)(q3 >> 32);
  return __builtin_bit_cast(v16bf, au);
}

// ------------------------------- main scan -----------------------------------
__global__ __launch_bounds__(256) void snn_scan_kernel(
    const unsigned int* __restrict__ planes,           // [SEQ][B][KWORDS]
    const unsigned int* __restrict__ whi, const unsigned int* __restrict__ wlo, // [128][16][32][8]
    const unsigned int* __restrict__ rhi, const unsigned int* __restrict__ rlo, // [8][16][32][8]
    const unsigned int* __restrict__ ohi, const unsigned int* __restrict__ olo, // [8][1][32][8]
    float* __restrict__ out) {
  __shared__ unsigned short zbuf[ROWS * HDIM];       // z (bf16), 16 KB
  __shared__ float mbuf[ROWS * ADIM];                // 2 KB
  __shared__ unsigned long long lutA[16];            // nibble -> bf16x4

  const int tid  = threadIdx.x;
  const int lane = tid & 31;
  const int wave = tid >> 5;
  const int bid  = blockIdx.x;
  const int half = lane >> 4;
  const int lrow = lane & 15;
  const int n0 = wave * 2, n1 = wave * 2 + 1;
  const int nb0 = n0 * 16, nb1 = n1 * 16;

  if (tid < 16) {
    unsigned n = (unsigned)tid;
    unsigned d0 = ((n)      & 1u) * 0x3F80u | ((n >> 1) & 1u) * 0x3F800000u;
    unsigned d1 = ((n >> 2) & 1u) * 0x3F80u | ((n >> 3) & 1u) * 0x3F800000u;
    lutA[tid] = (unsigned long long)d0 | ((unsigned long long)d1 << 32);
  }
  for (int i = tid; i < ROWS * HDIM; i += 256) zbuf[i] = 0;   // z0 = 0
  __syncthreads();

  const int rowG0 = bid * ROWS + lrow;        // M-tile 0 row of this lane
  const int rowG1 = rowG0 + 16;               // M-tile 1 row

  v8f v[2][2] = {}, ic[2][2] = {};            // LIF state, C-fragment layout
  v8f vo = {}, io = {}, m;                    // readout state (waves 0,1)
#pragma unroll
  for (int j = 0; j < 8; ++j) m[j] = -__builtin_inff();

  for (int t = 0; t < SEQ; ++t) {
    v8f acc[2][2] = {};

    // --- recurrent term: z_old @ w_rec.T  (K=256) ---
    for (int kb = 0; kb < HDIM / 32; ++kb) {
      v16bf am0 = load_z_afrag(zbuf, kb * 32, lrow, half);
      v16bf am1 = load_z_afrag(zbuf + 16 * HDIM, kb * 32, lrow, half);
      u32x8 bh0 = *(const u32x8*)(rhi + frag_off(kb, n0, 16, lane));
      u32x8 bl0 = *(const u32x8*)(rlo + frag_off(kb, n0, 16, lane));
      u32x8 bh1 = *(const u32x8*)(rhi + frag_off(kb, n1, 16, lane));
      u32x8 bl1 = *(const u32x8*)(rlo + frag_off(kb, n1, 16, lane));
      acc[0][0] = wmma_bf16(am0, bh0, acc[0][0]); acc[0][0] = wmma_bf16(am0, bl0, acc[0][0]);
      acc[0][1] = wmma_bf16(am0, bh1, acc[0][1]); acc[0][1] = wmma_bf16(am0, bl1, acc[0][1]);
      acc[1][0] = wmma_bf16(am1, bh0, acc[1][0]); acc[1][0] = wmma_bf16(am1, bl0, acc[1][0]);
      acc[1][1] = wmma_bf16(am1, bh1, acc[1][1]); acc[1][1] = wmma_bf16(am1, bl1, acc[1][1]);
    }

    // --- input term: enc_t @ w_in.T  (K=4096, A from bitplanes via LUT) ---
    const unsigned int* pr0 = planes + ((size_t)t * BDIM + rowG0) * KWORDS;
    const unsigned int* pr1 = planes + ((size_t)t * BDIM + rowG1) * KWORDS;
    for (int kb = 0; kb < KWORDS; ++kb) {
      v16bf am0 = expand_afrag(lutA, pr0[kb], half);
      v16bf am1 = expand_afrag(lutA, pr1[kb], half);
      u32x8 bh0 = *(const u32x8*)(whi + frag_off(kb, n0, 16, lane));
      u32x8 bl0 = *(const u32x8*)(wlo + frag_off(kb, n0, 16, lane));
      u32x8 bh1 = *(const u32x8*)(whi + frag_off(kb, n1, 16, lane));
      u32x8 bl1 = *(const u32x8*)(wlo + frag_off(kb, n1, 16, lane));
      acc[0][0] = wmma_bf16(am0, bh0, acc[0][0]); acc[0][0] = wmma_bf16(am0, bl0, acc[0][0]);
      acc[0][1] = wmma_bf16(am0, bh1, acc[0][1]); acc[0][1] = wmma_bf16(am0, bl1, acc[0][1]);
      acc[1][0] = wmma_bf16(am1, bh0, acc[1][0]); acc[1][0] = wmma_bf16(am1, bl0, acc[1][0]);
      acc[1][1] = wmma_bf16(am1, bh1, acc[1][1]); acc[1][1] = wmma_bf16(am1, bl1, acc[1][1]);
    }

    // --- LIF state update (elementwise on C fragments) ---
    unsigned zm[2][2] = {{0u, 0u}, {0u, 0u}};
#pragma unroll
    for (int mt = 0; mt < 2; ++mt) {
#pragma unroll
      for (int nt = 0; nt < 2; ++nt) {
#pragma unroll
        for (int j = 0; j < 8; ++j) {
          float vd = v[mt][nt][j] + 0.1f * ((0.0f - v[mt][nt][j]) + ic[mt][nt][j]);
          float id = ic[mt][nt][j] * 0.8f;
          float z  = (vd - 1.0f) > 0.0f ? 1.0f : 0.0f;
          v[mt][nt][j]  = (1.0f - z) * vd;
          ic[mt][nt][j] = id + acc[mt][nt][j];
          zm[mt][nt] |= (z > 0.0f) ? (1u << j) : 0u;
        }
      }
    }

    __syncthreads();                 // all waves done reading z_old
#pragma unroll
    for (int mt = 0; mt < 2; ++mt) {
#pragma unroll
      for (int j = 0; j < 8; ++j) {  // publish z_new (bf16 0/1)
        int M = mt * 16 + j + 8 * half;
        zbuf[M * HDIM + nb0 + lrow] = ((zm[mt][0] >> j) & 1u) ? (unsigned short)0x3F80 : 0;
        zbuf[M * HDIM + nb1 + lrow] = ((zm[mt][1] >> j) & 1u) ? (unsigned short)0x3F80 : 0;
      }
    }
    __syncthreads();                 // z_new visible

    // --- LI readout: waves 0,1 each handle one 16-row M-tile ---
    if (wave < 2) {
      v8f accO = {};
      const unsigned short* zmt = zbuf + wave * 16 * HDIM;
      for (int kb = 0; kb < HDIM / 32; ++kb) {
        v16bf a = load_z_afrag(zmt, kb * 32, lrow, half);
        u32x8 bh = *(const u32x8*)(ohi + frag_off(kb, 0, 1, lane));
        u32x8 bl = *(const u32x8*)(olo + frag_off(kb, 0, 1, lane));
        accO = wmma_bf16(a, bh, accO);
        accO = wmma_bf16(a, bl, accO);
      }
#pragma unroll
      for (int j = 0; j < 8; ++j) {
        float von = vo[j] + 0.1f * ((0.0f - vo[j]) + io[j]);  // uses old io
        io[j] = io[j] * 0.8f + accO[j];
        vo[j] = von;
        m[j]  = fmaxf(m[j], von);
      }
    }
  }

  __syncthreads();
  if (wave < 2) {
#pragma unroll
    for (int j = 0; j < 8; ++j) {
      float val = m[j];
      val = (val != val) ? 0.0f : val;               // NaN -> 0
      mbuf[(wave * 16 + j + 8 * half) * ADIM + lrow] = val;
    }
  }
  __syncthreads();

  if (tid < ROWS) {                                   // softmax over A=16
    float row[ADIM];
    float mx = -__builtin_inff();
    for (int a = 0; a < ADIM; ++a) { row[a] = mbuf[tid * ADIM + a]; mx = fmaxf(mx, row[a]); }
    float s = 0.0f;
    for (int a = 0; a < ADIM; ++a) { row[a] = expf(row[a] - mx); s += row[a]; }
    float inv = 1.0f / s;
    for (int a = 0; a < ADIM; ++a)
      out[((size_t)bid * ROWS + tid) * ADIM + a] = row[a] * inv;
  }
}

// ------------------------------- launcher ------------------------------------
extern "C" void kernel_launch(void* const* d_in, const int* in_sizes, int n_in,
                              void* d_out, int out_size, void* d_ws, size_t ws_size,
                              hipStream_t stream) {
  const float* x     = (const float*)d_in[0];  // (2048, 2048)
  const float* w_in  = (const float*)d_in[1];  // (256, 4096)
  const float* w_rec = (const float*)d_in[2];  // (256, 256)
  const float* w_out = (const float*)d_in[3];  // (16, 256)
  float* out = (float*)d_out;

  // Workspace layout (bytes); total ~14.3 MB.
  char* ws = (char*)d_ws;
  const size_t SZ_PLANES = (size_t)SEQ * BDIM * KWORDS * 4;            // 10 MB
  const size_t SZ_WIN    = (size_t)(KIN / 32) * 16 * 32 * 8 * 4;       // 2 MB
  const size_t SZ_REC    = (size_t)(HDIM / 32) * 16 * 32 * 8 * 4;      // 128 KB
  const size_t SZ_OUT    = (size_t)(HDIM / 32) * 1 * 32 * 8 * 4;       // 8 KB
  const size_t OFF_PLANES = 0;
  const size_t OFF_WHI = OFF_PLANES + SZ_PLANES;
  const size_t OFF_WLO = OFF_WHI + SZ_WIN;
  const size_t OFF_RHI = OFF_WLO + SZ_WIN;
  const size_t OFF_RLO = OFF_RHI + SZ_REC;
  const size_t OFF_OHI = OFF_RLO + SZ_REC;
  const size_t OFF_OLO = OFF_OHI + SZ_OUT;

  unsigned int* planes = (unsigned int*)(ws + OFF_PLANES);
  unsigned int* whi    = (unsigned int*)(ws + OFF_WHI);
  unsigned int* wlo    = (unsigned int*)(ws + OFF_WLO);
  unsigned int* rhi    = (unsigned int*)(ws + OFF_RHI);
  unsigned int* rlo    = (unsigned int*)(ws + OFF_RLO);
  unsigned int* ohi    = (unsigned int*)(ws + OFF_OHI);
  unsigned int* olo    = (unsigned int*)(ws + OFF_OLO);

  const int nWords = BDIM * (SDIM / 32);           // 131072
  encode_planes_kernel<<<(nWords + 255) / 256, 256, 0, stream>>>(x, planes, nWords);

  const int tot_win = (KIN / 32) * 16 * 32 * 8;    // 524288
  const int tot_rec = (HDIM / 32) * 16 * 32 * 8;   // 32768
  const int tot_out = (HDIM / 32) * 1 * 32 * 8;    // 2048
  pack_frag_kernel<<<(tot_win + 255) / 256, 256, 0, stream>>>(w_in, KIN, 16, whi, wlo, tot_win);
  pack_frag_kernel<<<(tot_rec + 255) / 256, 256, 0, stream>>>(w_rec, HDIM, 16, rhi, rlo, tot_rec);
  pack_frag_kernel<<<(tot_out + 255) / 256, 256, 0, stream>>>(w_out, HDIM, 1, ohi, olo, tot_out);

  snn_scan_kernel<<<BDIM / ROWS, 256, 0, stream>>>(planes, whi, wlo, rhi, rlo, ohi, olo, out);
}
